// MoEConformerLayer_33990371181313
// MI455X (gfx1250) — compile-verified, matrix operand order checked by reading, not run on hardware
//
#include <hip/hip_runtime.h>

// ---------------------------------------------------------------------------
// MoE Conformer layer for gfx1250 (MI455X), wave32 + WMMA bf16.
// B=2 S=2048 D=512 H=8 HD=64 K=31 FF=2048 G=4 E=2, N = B*S = 4096.
// ---------------------------------------------------------------------------

typedef unsigned short u16;
typedef __attribute__((ext_vector_type(16))) __bf16 v16bf;
typedef __attribute__((ext_vector_type(16))) unsigned short v16u;
typedef __attribute__((ext_vector_type(8)))  float  v8f;
typedef __attribute__((ext_vector_type(4)))  int    v4i;

#define NTOK 4096
#define DIM  512
#define SEQ  2048
#define FFD  2048

#define ASTRIDE 40   // 32 + 8 pad (u16); 80B row pitch (16B aligned)
#define QSTRIDE 72   // 64 + 8 pad (u16); 144B row pitch (16B aligned)

// ---- async global->LDS copy (CDNA5), with safe fallback -------------------
#if defined(__has_builtin)
#if __has_builtin(__builtin_amdgcn_global_load_async_to_lds_b128) && \
    __has_builtin(__builtin_amdgcn_s_wait_asynccnt)
#define USE_ASYNC_LDS 1
#endif
#endif

#ifdef USE_ASYNC_LDS
__device__ __forceinline__ void cp16(u16* l, const u16* g) {
  __builtin_amdgcn_global_load_async_to_lds_b128((v4i*)g, (v4i*)l, 0, 0);
}
__device__ __forceinline__ void cp_wait() { __builtin_amdgcn_s_wait_asynccnt(0); }
#else
__device__ __forceinline__ void cp16(u16* l, const u16* g) {
  *(uint4*)l = *(const uint4*)g;
}
__device__ __forceinline__ void cp_wait() {}
#endif

// ---- scalar helpers -------------------------------------------------------

__device__ __forceinline__ u16 f2bf(float f) {
  unsigned u = __builtin_bit_cast(unsigned, f);
  u += 0x7FFFu + ((u >> 16) & 1u);       // round-to-nearest-even
  return (u16)(u >> 16);
}

// tanh-form GELU via exact logistic identity: branch-free (v_exp + v_rcp).
__device__ __forceinline__ float gelu_f(float x) {
  float x3 = x * x * x;
  float t = __expf(-1.5957691216f * (x + 0.044715f * x3));
  return x * __builtin_amdgcn_rcpf(1.0f + t);
}

// ---- WMMA fragment helper (CDNA5 ISA 7.12.2) ------------------------------
// A 16x32 bf16: lane L<16 holds row M=L: elems[0..7]=K 0..7, [8..15]=K 16..23
//               lane L+16:   row M=L:   elems[0..7]=K 8..15, [8..15]=K 24..31
// B operand (32x16) has the mirrored mapping (lane = column), so a B tile
// staged TRANSPOSED ([N][K]) is loaded with this exact same function.
__device__ __forceinline__ v16bf afrag(const u16* p, int stride) {
  int lane = threadIdx.x & 31;
  int m = lane & 15, h = (lane >> 4) & 1;
  const u16* r = p + m * stride + h * 8;
  v16u t;
#pragma unroll
  for (int i = 0; i < 8; ++i) { t[i] = r[i]; t[i + 8] = r[16 + i]; }
  return __builtin_bit_cast(v16bf, t);
}

__device__ __forceinline__ v8f wmma_bf16(v16bf a, v16bf b, v8f c) {
  return __builtin_amdgcn_wmma_f32_16x16x32_bf16(
      false, a, false, b, (short)0, c, false, false);
}

// One K=32 step of a 128x64 block tile; wave computes 2x2 16x16 frags.
// la: [128][ASTRIDE] rows=M.  lb: [64][ASTRIDE] rows=N (transposed B).
__device__ __forceinline__ void mma_step(const u16* la, const u16* lb,
                                         v8f acc[2][2], int wrow, int wcol) {
  v16bf a0 = afrag(la + (wrow)      * ASTRIDE, ASTRIDE);
  v16bf a1 = afrag(la + (wrow + 16) * ASTRIDE, ASTRIDE);
  v16bf b0 = afrag(lb + (wcol)      * ASTRIDE, ASTRIDE);
  v16bf b1 = afrag(lb + (wcol + 16) * ASTRIDE, ASTRIDE);
  acc[0][0] = wmma_bf16(a0, b0, acc[0][0]);
  acc[0][1] = wmma_bf16(a0, b1, acc[0][1]);
  acc[1][0] = wmma_bf16(a1, b0, acc[1][0]);
  acc[1][1] = wmma_bf16(a1, b1, acc[1][1]);
}

// ---------------------------------------------------------------------------
// LayerNorm row (512) -> bf16.  grid = NTOK, block = 128.
// ---------------------------------------------------------------------------
__global__ void ln_kernel(const float* __restrict__ x,
                          const float* __restrict__ sc,
                          const float* __restrict__ bi,
                          u16* __restrict__ out) {
  int row = blockIdx.x, tid = threadIdx.x;
  const float4 v = ((const float4*)(x + row * DIM))[tid];
  float s  = v.x + v.y + v.z + v.w;
  float ss = v.x * v.x + v.y * v.y + v.z * v.z + v.w * v.w;
#pragma unroll
  for (int o = 16; o; o >>= 1) { s += __shfl_xor(s, o); ss += __shfl_xor(ss, o); }
  __shared__ float red[8];
  int w = tid >> 5;
  if ((tid & 31) == 0) { red[w] = s; red[4 + w] = ss; }
  __syncthreads();
  float t = 0.f, t2 = 0.f;
#pragma unroll
  for (int i = 0; i < 4; ++i) { t += red[i]; t2 += red[4 + i]; }
  float mean = t * (1.0f / DIM);
  float var  = t2 * (1.0f / DIM) - mean * mean;
  float rs   = rsqrtf(var + 1e-6f);
  int c = tid * 4;
  float xv[4] = {v.x, v.y, v.z, v.w};
#pragma unroll
  for (int j = 0; j < 4; ++j)
    out[row * DIM + c + j] = f2bf((xv[j] - mean) * rs * sc[c + j] + bi[c + j]);
}

// ---------------------------------------------------------------------------
// conv1d(K=31, SAME) as 31 shifted GEMMs + bias + gelu + residual.
// grid = (DIM/64, NTOK/128), block = 256.
// ---------------------------------------------------------------------------
__global__ void conv_kernel(const u16* __restrict__ h1,
                            const float* __restrict__ ck,
                            const float* __restrict__ cb,
                            const float* __restrict__ x0,
                            float* __restrict__ x1) {
  __shared__ u16 la[128 * ASTRIDE];
  __shared__ u16 lb[64 * ASTRIDE];
  int tid = threadIdx.x, lane = tid & 31, w = tid >> 5;
  int wrow = (w & 3) * 32, wcol = (w >> 2) * 32;
  int mbase = blockIdx.y * 128, nbase = blockIdx.x * 64;
  v8f acc[2][2] = {};
  for (int k = 0; k < 31; ++k) {
    for (int dk = 0; dk < 16; ++dk) {
      __syncthreads();
      // A: 128x32 bf16 copy (shifted rows), 16B chunks
      for (int t = tid; t < 128 * 4; t += 256) {
        int i = t >> 2, jc = (t & 3) * 8;
        int n = mbase + i, s = n & (SEQ - 1);
        int sp = s + k - 15;
        u16* dst = la + i * ASTRIDE + jc;
        if ((unsigned)sp < (unsigned)SEQ)
          cp16(dst, h1 + (n - s + sp) * DIM + dk * 32 + jc);
        else
          *(uint4*)dst = uint4{0, 0, 0, 0};
      }
      // B transposed: lb[n][kk] = ck[k, dk*32+kk, nbase+n]
      __builtin_prefetch(ck + (k * DIM + dk * 32 + 32 + (tid & 31)) * DIM + nbase, 0, 1);
      for (int t = tid; t < 64 * 32; t += 256) {
        int j = t & 63, kk = t >> 6;
        lb[j * ASTRIDE + kk] = f2bf(ck[(k * DIM + dk * 32 + kk) * DIM + nbase + j]);
      }
      cp_wait();
      __syncthreads();
      mma_step(la, lb, acc, wrow, wcol);
    }
  }
  int half = lane >> 4, nlo = lane & 15;
#pragma unroll
  for (int mi = 0; mi < 2; ++mi)
#pragma unroll
    for (int ni = 0; ni < 2; ++ni)
#pragma unroll
      for (int e = 0; e < 8; ++e) {
        int n = mbase + wrow + mi * 16 + e + 8 * half;
        int c = nbase + wcol + ni * 16 + nlo;
        float v = acc[mi][ni][e] + cb[c];
        x1[n * DIM + c] = gelu_f(v) + x0[n * DIM + c];
      }
}

// ---------------------------------------------------------------------------
// Projection GEMM [N,512]x[512,512]+bias -> bf16 in [B,H,S,HD] (q scaled).
// grid = (8, 32), block = 256.
// ---------------------------------------------------------------------------
__global__ void proj_kernel(const u16* __restrict__ a,
                            const float* __restrict__ wgt,
                            const float* __restrict__ bias,
                            u16* __restrict__ out, float scale) {
  __shared__ u16 la[128 * ASTRIDE];
  __shared__ u16 lb[64 * ASTRIDE];
  int tid = threadIdx.x, lane = tid & 31, w = tid >> 5;
  int wrow = (w & 3) * 32, wcol = (w >> 2) * 32;
  int mbase = blockIdx.y * 128, nbase = blockIdx.x * 64;
  v8f acc[2][2] = {};
  for (int dk = 0; dk < 16; ++dk) {
    __syncthreads();
    for (int t = tid; t < 128 * 4; t += 256) {
      int i = t >> 2, jc = (t & 3) * 8;
      cp16(la + i * ASTRIDE + jc, a + (mbase + i) * DIM + dk * 32 + jc);
    }
    for (int t = tid; t < 64 * 32; t += 256) {
      int j = t & 63, kk = t >> 6;
      lb[j * ASTRIDE + kk] = f2bf(wgt[(dk * 32 + kk) * DIM + nbase + j]);
    }
    cp_wait();
    __syncthreads();
    mma_step(la, lb, acc, wrow, wcol);
  }
  int half = lane >> 4, nlo = lane & 15;
#pragma unroll
  for (int mi = 0; mi < 2; ++mi)
#pragma unroll
    for (int ni = 0; ni < 2; ++ni)
#pragma unroll
      for (int e = 0; e < 8; ++e) {
        int n = mbase + wrow + mi * 16 + e + 8 * half;
        int c = nbase + wcol + ni * 16 + nlo;
        int b = n >> 11, s = n & (SEQ - 1);
        int h = c >> 6, hd = c & 63;
        float v = (acc[mi][ni][e] + bias[c]) * scale;
        out[((b * 8 + h) * SEQ + s) * 64 + hd] = f2bf(v);
      }
}

// ---------------------------------------------------------------------------
// Flash attention (non-causal), online softmax over S.
// grid = (B*H, S/128), block = 256 (8 waves x 16 q-rows).
// K tile consumed in natural [s][hd] layout (B-operand == transposed stage);
// V tile staged transposed [hd][s].
// ---------------------------------------------------------------------------
__global__ void attn_kernel(const u16* __restrict__ qb,
                            const u16* __restrict__ kb,
                            const u16* __restrict__ vb,
                            u16* __restrict__ ob) {
  __shared__ u16 sm[18432];                 // 36 KB
  u16* lq  = sm;                            // [128][72] (transient)
  u16* lk  = sm;                            // [64][72]  K rows: [kcol][hd]
  u16* lv  = sm + 64 * QSTRIDE;             // [64][72]  V^T:    [hd][srow]
  u16* lpb = sm + 128 * QSTRIDE;            // 8 waves x [16][72]
  int tid = threadIdx.x, lane = tid & 31, w = tid >> 5;
  int half = lane >> 4, nlo = lane & 15;
  int bh = blockIdx.x, qt = blockIdx.y;

  for (int t = tid; t < 128 * 8; t += 256) {
    int i = t >> 3, jc = (t & 7) * 8;
    cp16(lq + i * QSTRIDE + jc, qb + (bh * SEQ + qt * 128 + i) * 64 + jc);
  }
  cp_wait();
  __syncthreads();
  v16bf qa0 = afrag(lq + (w * 16) * QSTRIDE, QSTRIDE);       // hd 0..31
  v16bf qa1 = afrag(lq + (w * 16) * QSTRIDE + 32, QSTRIDE);  // hd 32..63
  __syncthreads();

  float mrun[8], lrun[8];
  v8f oacc[4] = {};
#pragma unroll
  for (int e = 0; e < 8; ++e) { mrun[e] = -3.0e38f; lrun[e] = 0.f; }

  for (int kt = 0; kt < SEQ / 64; ++kt) {
    // K tile: straight row copy (async-capable)
    for (int t = tid; t < 64 * 8; t += 256) {
      int i = t >> 3, jc = (t & 7) * 8;
      cp16(lk + i * QSTRIDE + jc, kb + (bh * SEQ + kt * 64 + i) * 64 + jc);
    }
    // V tile: transpose while staging
    for (int t = tid; t < 64 * 64; t += 256) {
      int i = t >> 6, j = t & 63;  // i = srow, j = hd
      lv[j * QSTRIDE + i] = vb[(bh * SEQ + kt * 64 + i) * 64 + j];
    }
    cp_wait();
    __syncthreads();

    v8f sf[4];
#pragma unroll
    for (int ni = 0; ni < 4; ++ni) {
      v8f z = {};
      sf[ni] = wmma_bf16(qa0, afrag(lk + (ni * 16) * QSTRIDE,      QSTRIDE), z);
      sf[ni] = wmma_bf16(qa1, afrag(lk + (ni * 16) * QSTRIDE + 32, QSTRIDE), sf[ni]);
    }

    float mnew[8], fac[8], rs[8];
#pragma unroll
    for (int e = 0; e < 8; ++e) {
      float mx = fmaxf(fmaxf(sf[0][e], sf[1][e]), fmaxf(sf[2][e], sf[3][e]));
#pragma unroll
      for (int o = 1; o < 16; o <<= 1) mx = fmaxf(mx, __shfl_xor(mx, o));
      mnew[e] = fmaxf(mrun[e], mx);
      fac[e]  = __expf(mrun[e] - mnew[e]);
      mrun[e] = mnew[e];
      rs[e]   = 0.f;
    }
#pragma unroll
    for (int ni = 0; ni < 4; ++ni)
#pragma unroll
      for (int e = 0; e < 8; ++e) {
        float p = __expf(sf[ni][e] - mnew[e]);
        sf[ni][e] = p;
        rs[e] += p;
      }
#pragma unroll
    for (int e = 0; e < 8; ++e) {
      float r = rs[e];
#pragma unroll
      for (int o = 1; o < 16; o <<= 1) r += __shfl_xor(r, o);
      lrun[e] = lrun[e] * fac[e] + r;
    }
#pragma unroll
    for (int ni = 0; ni < 4; ++ni)
#pragma unroll
      for (int e = 0; e < 8; ++e) oacc[ni][e] *= fac[e];

    u16* lp = lpb + w * 16 * QSTRIDE;
#pragma unroll
    for (int ni = 0; ni < 4; ++ni)
#pragma unroll
      for (int e = 0; e < 8; ++e)
        lp[(e + 8 * half) * QSTRIDE + ni * 16 + nlo] = f2bf(sf[ni][e]);
    __syncthreads();

    v16bf pa0 = afrag(lp, QSTRIDE);        // s 0..31
    v16bf pa1 = afrag(lp + 32, QSTRIDE);   // s 32..63
#pragma unroll
    for (int ni = 0; ni < 4; ++ni) {
      oacc[ni] = wmma_bf16(pa0, afrag(lv + (ni * 16) * QSTRIDE,      QSTRIDE), oacc[ni]);
      oacc[ni] = wmma_bf16(pa1, afrag(lv + (ni * 16) * QSTRIDE + 32, QSTRIDE), oacc[ni]);
    }
    __syncthreads();
  }

  int b = bh >> 3, h = bh & 7;
#pragma unroll
  for (int e = 0; e < 8; ++e) {
    float inv = __builtin_amdgcn_rcpf(lrun[e]);
    int row = qt * 128 + w * 16 + e + 8 * half;
#pragma unroll
    for (int ni = 0; ni < 4; ++ni) {
      int hc = h * 64 + ni * 16 + nlo;
      ob[(b * SEQ + row) * DIM + hc] = f2bf(oacc[ni][e] * inv);
    }
  }
}

// ---------------------------------------------------------------------------
// Output projection + residual -> x2 (f32) and x2 bf16 copy.
// grid = (8, 32), block = 256.
// ---------------------------------------------------------------------------
__global__ void oproj_kernel(const u16* __restrict__ a,
                             const float* __restrict__ wo,
                             const float* __restrict__ bo,
                             const float* __restrict__ res,
                             float* __restrict__ x2,
                             u16* __restrict__ x2b) {
  __shared__ u16 la[128 * ASTRIDE];
  __shared__ u16 lb[64 * ASTRIDE];
  int tid = threadIdx.x, lane = tid & 31, w = tid >> 5;
  int wrow = (w & 3) * 32, wcol = (w >> 2) * 32;
  int mbase = blockIdx.y * 128, nbase = blockIdx.x * 64;
  v8f acc[2][2] = {};
  for (int dk = 0; dk < 16; ++dk) {
    __syncthreads();
    for (int t = tid; t < 128 * 4; t += 256) {
      int i = t >> 2, jc = (t & 3) * 8;
      cp16(la + i * ASTRIDE + jc, a + (mbase + i) * DIM + dk * 32 + jc);
    }
    for (int t = tid; t < 64 * 32; t += 256) {
      int j = t & 63, kk = t >> 6;
      lb[j * ASTRIDE + kk] = f2bf(wo[(dk * 32 + kk) * DIM + nbase + j]);
    }
    cp_wait();
    __syncthreads();
    mma_step(la, lb, acc, wrow, wcol);
  }
  int half = lane >> 4, nlo = lane & 15;
#pragma unroll
  for (int mi = 0; mi < 2; ++mi)
#pragma unroll
    for (int ni = 0; ni < 2; ++ni)
#pragma unroll
      for (int e = 0; e < 8; ++e) {
        int n = mbase + wrow + mi * 16 + e + 8 * half;
        int c = nbase + wcol + ni * 16 + nlo;
        float v = acc[mi][ni][e] + bo[c] + res[n * DIM + c];
        x2[n * DIM + c]  = v;
        x2b[n * DIM + c] = f2bf(v);
      }
}

// ---------------------------------------------------------------------------
// MoE routing (group bucketing).
// ---------------------------------------------------------------------------
__global__ void zero_counts_kernel(int* counts) {
  if (threadIdx.x < 4) counts[threadIdx.x] = 0;
}
__global__ void route_kernel(const int* __restrict__ gid,
                             int* __restrict__ counts,
                             int* __restrict__ order) {
  int n = blockIdx.x * 256 + threadIdx.x;
  int g = gid[n];
  int pos = atomicAdd(&counts[g], 1);
  order[g * NTOK + pos] = n;
}

// ---------------------------------------------------------------------------
// MoE FFN stage 1: hid[e][n,ff] = gelu(x @ w1[g,e] + b1) (bf16, routed).
// grid = (FF/64, N/128, G*E), block = 256.
// ---------------------------------------------------------------------------
__global__ void moe1_kernel(const u16* __restrict__ x2b,
                            const float* __restrict__ w1,
                            const float* __restrict__ b1,
                            const int* __restrict__ counts,
                            const int* __restrict__ order,
                            u16* __restrict__ hid) {
  __shared__ u16 la[128 * ASTRIDE];
  __shared__ u16 lb[64 * ASTRIDE];
  int ge = blockIdx.z, g = ge >> 1, e = ge & 1;
  int cnt = counts[g];
  int mbase = blockIdx.y * 128;
  if (mbase >= cnt) return;
  int nbase = blockIdx.x * 64;
  int tid = threadIdx.x, lane = tid & 31, w = tid >> 5;
  int wrow = (w & 3) * 32, wcol = (w >> 2) * 32;
  const int* ord = order + g * NTOK;
  v8f acc[2][2] = {};
  for (int dk = 0; dk < 16; ++dk) {
    __syncthreads();
    for (int t = tid; t < 128 * 4; t += 256) {
      int i = t >> 2, jc = (t & 3) * 8;
      int pos = mbase + i;
      u16* dst = la + i * ASTRIDE + jc;
      if (pos < cnt) cp16(dst, x2b + ord[pos] * DIM + dk * 32 + jc);
      else           *(uint4*)dst = uint4{0, 0, 0, 0};
    }
    __builtin_prefetch(w1 + (ge * DIM + dk * 32 + 32 + (tid & 31)) * FFD + nbase, 0, 1);
    for (int t = tid; t < 64 * 32; t += 256) {
      int j = t & 63, kk = t >> 6;
      lb[j * ASTRIDE + kk] = f2bf(w1[(ge * DIM + dk * 32 + kk) * FFD + nbase + j]);
    }
    cp_wait();
    __syncthreads();
    mma_step(la, lb, acc, wrow, wcol);
  }
  int half = lane >> 4, nlo = lane & 15;
#pragma unroll
  for (int mi = 0; mi < 2; ++mi)
#pragma unroll
    for (int ni = 0; ni < 2; ++ni)
#pragma unroll
      for (int ee = 0; ee < 8; ++ee) {
        int pos = mbase + wrow + mi * 16 + ee + 8 * half;
        if (pos < cnt) {
          int n = ord[pos];
          int c = nbase + wcol + ni * 16 + nlo;
          float v = gelu_f(acc[mi][ni][ee] + b1[ge * FFD + c]);
          hid[(e * NTOK + n) * FFD + c] = f2bf(v);
        }
      }
}

// ---------------------------------------------------------------------------
// MoE FFN stage 2: y = 0.5*sum_e(hid_e @ w2[g,e]) + mean(b2) + res.
// grid = (D/64, N/128, G), block = 256.
// ---------------------------------------------------------------------------
__global__ void moe2_kernel(const u16* __restrict__ hid,
                            const float* __restrict__ w2,
                            const float* __restrict__ b2,
                            const int* __restrict__ counts,
                            const int* __restrict__ order,
                            const float* __restrict__ x2,
                            float* __restrict__ out) {
  __shared__ u16 la[128 * ASTRIDE];
  __shared__ u16 lb[64 * ASTRIDE];
  int g = blockIdx.z;
  int cnt = counts[g];
  int mbase = blockIdx.y * 128;
  if (mbase >= cnt) return;
  int nbase = blockIdx.x * 64;
  int tid = threadIdx.x, lane = tid & 31, w = tid >> 5;
  int wrow = (w & 3) * 32, wcol = (w >> 2) * 32;
  const int* ord = order + g * NTOK;
  v8f acc[2][2] = {};
  for (int e = 0; e < 2; ++e) {
    const u16* he = hid + e * NTOK * FFD;
    const float* w2e = w2 + (g * 2 + e) * FFD * DIM;
    for (int kc = 0; kc < FFD / 32; ++kc) {
      __syncthreads();
      for (int t = tid; t < 128 * 4; t += 256) {
        int i = t >> 2, jc = (t & 3) * 8;
        int pos = mbase + i;
        u16* dst = la + i * ASTRIDE + jc;
        if (pos < cnt) cp16(dst, he + ord[pos] * FFD + kc * 32 + jc);
        else           *(uint4*)dst = uint4{0, 0, 0, 0};
      }
      __builtin_prefetch(w2e + (kc * 32 + 32 + (tid & 31)) * DIM + nbase, 0, 1);
      for (int t = tid; t < 64 * 32; t += 256) {
        int j = t & 63, kk = t >> 6;
        lb[j * ASTRIDE + kk] = f2bf(w2e[(kc * 32 + kk) * DIM + nbase + j]);
      }
      cp_wait();
      __syncthreads();
      mma_step(la, lb, acc, wrow, wcol);
    }
  }
  int half = lane >> 4, nlo = lane & 15;
#pragma unroll
  for (int mi = 0; mi < 2; ++mi)
#pragma unroll
    for (int ni = 0; ni < 2; ++ni)
#pragma unroll
      for (int ee = 0; ee < 8; ++ee) {
        int pos = mbase + wrow + mi * 16 + ee + 8 * half;
        if (pos < cnt) {
          int n = ord[pos];
          int c = nbase + wcol + ni * 16 + nlo;
          float bmean = 0.5f * (b2[(g * 2 + 0) * DIM + c] + b2[(g * 2 + 1) * DIM + c]);
          out[n * DIM + c] = 0.5f * acc[mi][ni][ee] + bmean + x2[n * DIM + c];
        }
      }
}

// ---------------------------------------------------------------------------
// Launch
// ---------------------------------------------------------------------------
extern "C" void kernel_launch(void* const* d_in, const int* in_sizes, int n_in,
                              void* d_out, int out_size, void* d_ws, size_t ws_size,
                              hipStream_t stream) {
  const float* x    = (const float*)d_in[0];
  const int*   gid  = (const int*)d_in[1];
  const float* ln1s = (const float*)d_in[2];
  const float* ln1b = (const float*)d_in[3];
  const float* ck   = (const float*)d_in[4];
  const float* cb   = (const float*)d_in[5];
  const float* ln2s = (const float*)d_in[6];
  const float* ln2b = (const float*)d_in[7];
  const float* wq   = (const float*)d_in[8];
  const float* bq   = (const float*)d_in[9];
  const float* wk   = (const float*)d_in[10];
  const float* bk   = (const float*)d_in[11];
  const float* wv   = (const float*)d_in[12];
  const float* bv   = (const float*)d_in[13];
  const float* wo   = (const float*)d_in[14];
  const float* bo   = (const float*)d_in[15];
  const float* w1   = (const float*)d_in[16];
  const float* b1   = (const float*)d_in[17];
  const float* w2   = (const float*)d_in[18];
  const float* b2   = (const float*)d_in[19];
  float* out = (float*)d_out;

  // workspace carve-up (~77 MB)
  char* p = (char*)d_ws;
  u16*   h1   = (u16*)p;   p += (size_t)NTOK * DIM * 2;
  float* x1   = (float*)p; p += (size_t)NTOK * DIM * 4;
  u16*   h2   = (u16*)p;   p += (size_t)NTOK * DIM * 2;
  u16*   qb   = (u16*)p;   p += (size_t)NTOK * DIM * 2;
  u16*   kb   = (u16*)p;   p += (size_t)NTOK * DIM * 2;
  u16*   vb   = (u16*)p;   p += (size_t)NTOK * DIM * 2;
  u16*   ob   = (u16*)p;   p += (size_t)NTOK * DIM * 2;
  float* x2   = (float*)p; p += (size_t)NTOK * DIM * 4;
  u16*   x2b  = (u16*)p;   p += (size_t)NTOK * DIM * 2;
  u16*   hid  = (u16*)p;   p += (size_t)2 * NTOK * FFD * 2;
  int*   counts = (int*)p; p += 256;
  int*   order  = (int*)p; p += (size_t)4 * NTOK * 4;

  dim3 blk(256);

  // conv sublayer
  ln_kernel<<<NTOK, 128, 0, stream>>>(x, ln1s, ln1b, h1);
  conv_kernel<<<dim3(DIM / 64, NTOK / 128), blk, 0, stream>>>(h1, ck, cb, x, x1);

  // attention sublayer
  ln_kernel<<<NTOK, 128, 0, stream>>>(x1, ln2s, ln2b, h2);
  proj_kernel<<<dim3(DIM / 64, NTOK / 128), blk, 0, stream>>>(h2, wq, bq, qb, 0.125f);
  proj_kernel<<<dim3(DIM / 64, NTOK / 128), blk, 0, stream>>>(h2, wk, bk, kb, 1.0f);
  proj_kernel<<<dim3(DIM / 64, NTOK / 128), blk, 0, stream>>>(h2, wv, bv, vb, 1.0f);
  attn_kernel<<<dim3(16, SEQ / 128), blk, 0, stream>>>(qb, kb, vb, ob);
  oproj_kernel<<<dim3(DIM / 64, NTOK / 128), blk, 0, stream>>>(ob, wo, bo, x1, x2, x2b);

  // MoE sublayer (routed: only each token's own group's 2 experts)
  zero_counts_kernel<<<1, 32, 0, stream>>>(counts);
  route_kernel<<<NTOK / 256, 256, 0, stream>>>(gid, counts, order);
  moe1_kernel<<<dim3(FFD / 64, NTOK / 128, 8), blk, 0, stream>>>(x2b, w1, b1, counts, order, hid);
  moe2_kernel<<<dim3(DIM / 64, NTOK / 128, 4), blk, 0, stream>>>(hid, w2, b2, counts, order, x2, out);
}
